// CortexMLSTM_39848706573660
// MI455X (gfx1250) — matrix operand order repaired and checked by reading, not compiled, not run on hardware
//
#include <hip/hip_runtime.h>

// ---------------- problem constants ----------------
#define T_SEQ 512
#define NB    64
#define NIN   1024
#define NHID  1024
#define G4    4096          // 4*NHID
#define NWG2  16            // persistent workgroups for the scan kernel
#define LDS_PITCH 1032      // 1024 + 8 shorts pad -> 2064B row stride, conflict-free frags

typedef __attribute__((ext_vector_type(16))) __bf16 v16bf;
typedef __attribute__((ext_vector_type(8)))  __bf16 v8bf;
typedef __attribute__((ext_vector_type(8)))  float  v8f;
typedef __attribute__((ext_vector_type(4)))  float  v4f;
typedef __attribute__((ext_vector_type(8)))  unsigned short v8u16;

// ---------------- helpers ----------------
__device__ __forceinline__ unsigned short f2bf(float f) {
    unsigned u = __float_as_uint(f);
    u += 0x7FFFu + ((u >> 16) & 1u);       // round-to-nearest-even
    return (unsigned short)(u >> 16);
}

__device__ __forceinline__ v8f zero8() {
    v8f z = {0.f,0.f,0.f,0.f,0.f,0.f,0.f,0.f};
    return z;
}

__device__ __forceinline__ float sigmoid_f(float x) {
    return 1.0f / (1.0f + __expf(-x));
}
__device__ __forceinline__ float tanh_f(float x) {
    x = fminf(fmaxf(x, -15.0f), 15.0f);
    float e = __expf(2.0f * x);
    return (e - 1.0f) / (e + 1.0f);
}

union FragBF { v16bf v; v8bf h[2]; };

// A fragment, 16x32 bf16, row-major [*, ldk] in GLOBAL memory:
//   lanes 0-15: row = lane, K = {kh*8..+7} U {16+kh*8..+7}, kh = lane>>4
__device__ __forceinline__ v16bf load_frag_A(const unsigned short* __restrict__ base,
                                             int ldk, int row0, int kk, int lane) {
    int r  = lane & 15;
    int kh = lane >> 4;
    const unsigned short* p = base + (size_t)(row0 + r) * ldk + kk + kh * 8;
    FragBF u;
    u.h[0] = *reinterpret_cast<const v8bf*>(p);
    u.h[1] = *reinterpret_cast<const v8bf*>(p + 16);
    return u.v;
}

// A fragment sourced from padded LDS image of h (pitch = LDS_PITCH shorts)
__device__ __forceinline__ v16bf load_frag_A_lds(const unsigned short* hlds,
                                                 int row0, int kk, int lane) {
    int r  = lane & 15;
    int kh = lane >> 4;
    const unsigned short* p = hlds + (row0 + r) * LDS_PITCH + kk + kh * 8;
    FragBF u;
    u.h[0] = *reinterpret_cast<const v8bf*>(p);        // ds_load_b128
    u.h[1] = *reinterpret_cast<const v8bf*>(p + 16);   // ds_load_b128
    return u.v;
}

// B fragment, 32x16 bf16, source W row-major [N, ldk] (B(k,n)=W[n*ldk+k]):
//   lanes 0-15: col = lane, K = {kh*8..+7} U {16+kh*8..+7}, kh = lane>>4
__device__ __forceinline__ v16bf load_frag_B(const unsigned short* __restrict__ base,
                                             int ldk, int col0, int kk, int lane) {
    int c  = lane & 15;
    int kh = lane >> 4;
    const unsigned short* p = base + (size_t)(col0 + c) * ldk + kk + kh * 8;
    FragBF u;
    u.h[0] = *reinterpret_cast<const v8bf*>(p);
    u.h[1] = *reinterpret_cast<const v8bf*>(p + 16);
    return u.v;
}

// ---------------- f32 -> bf16 bulk convert (8 elems/thread) ----------------
__global__ __launch_bounds__(256) void cvt_f32_bf16(const float* __restrict__ src,
                                                    unsigned short* __restrict__ dst,
                                                    int n8) {
    int i = blockIdx.x * blockDim.x + threadIdx.x;
    if (i >= n8) return;
    const v4f* s = reinterpret_cast<const v4f*>(src) + (size_t)i * 2;
    v4f a = s[0], b = s[1];
    v8u16 o;
    o[0] = f2bf(a[0]); o[1] = f2bf(a[1]); o[2] = f2bf(a[2]); o[3] = f2bf(a[3]);
    o[4] = f2bf(b[0]); o[5] = f2bf(b[1]); o[6] = f2bf(b[2]); o[7] = f2bf(b[3]);
    reinterpret_cast<v8u16*>(dst)[i] = o;
}

// ---------------- phase 1: xg = x @ W_ih^T + (b_ih + b_hh) ----------------
// Wave computes 32(M) x 64(N): 8 accumulators; each B fragment feeds 2 WMMA.
// waves = (32768/32) * (4096/64) = 65536  ->  8192 blocks x 8 waves.
__global__ __launch_bounds__(256) void xgemm_kernel(const unsigned short* __restrict__ xbf,
                                                    const unsigned short* __restrict__ wbf,
                                                    const float* __restrict__ b_ih,
                                                    const float* __restrict__ b_hh,
                                                    float* __restrict__ xg) {
    const int lane = threadIdx.x & 31;
    const int wv   = blockIdx.x * 8 + (threadIdx.x >> 5);
    const int mBlk = wv >> 6;            // 0..1023  (32 rows each)
    const int nGrp = wv & 63;            // 0..63    (64 cols each)
    const int row0 = mBlk * 32;
    const int n0   = nGrp * 64;

    v8f acc[2][4];
#pragma unroll
    for (int s = 0; s < 2; ++s)
#pragma unroll
        for (int g = 0; g < 4; ++g) acc[s][g] = zero8();

    for (int kk = 0; kk < NIN; kk += 32) {
        if (kk + 32 < NIN)
            __builtin_prefetch(xbf + (size_t)(row0 + (lane & 15)) * NIN + kk + 32, 0, 1);

        v16bf a0 = load_frag_A(xbf, NIN, row0,      kk, lane);
        v16bf a1 = load_frag_A(xbf, NIN, row0 + 16, kk, lane);
#pragma unroll
        for (int g = 0; g < 4; ++g) {
            v16bf b = load_frag_B(wbf, NIN, n0 + g * 16, kk, lane);
            acc[0][g] = __builtin_amdgcn_wmma_f32_16x16x32_bf16(false, a0, false, b, (short)0, acc[0][g], false, false);
            acc[1][g] = __builtin_amdgcn_wmma_f32_16x16x32_bf16(false, a1, false, b, (short)0, acc[1][g], false, false);
        }
    }

    const int colLane = lane & 15;
    const int kh      = lane >> 4;
#pragma unroll
    for (int g = 0; g < 4; ++g) {
        const int n = n0 + g * 16 + colLane;
        const float bias = b_ih[n] + b_hh[n];
#pragma unroll
        for (int s = 0; s < 2; ++s)
#pragma unroll
            for (int r = 0; r < 8; ++r) {
                const int m = row0 + s * 16 + r + kh * 8;
                xg[(size_t)m * G4 + n] = acc[s][g][r] + bias;
            }
    }
}

// ---------------- grid-wide sense barrier ----------------
__device__ __forceinline__ void grid_barrier(unsigned* __restrict__ bar, unsigned nwg) {
    __syncthreads();
    if (threadIdx.x == 0) {
        __threadfence();
        unsigned gen0 = __hip_atomic_load(&bar[1], __ATOMIC_ACQUIRE, __HIP_MEMORY_SCOPE_AGENT);
        unsigned arrived = __hip_atomic_fetch_add(&bar[0], 1u, __ATOMIC_ACQ_REL, __HIP_MEMORY_SCOPE_AGENT);
        if (arrived == nwg - 1) {
            __hip_atomic_store(&bar[0], 0u, __ATOMIC_RELAXED, __HIP_MEMORY_SCOPE_AGENT);
            __hip_atomic_fetch_add(&bar[1], 1u, __ATOMIC_RELEASE, __HIP_MEMORY_SCOPE_AGENT);
        } else {
            while (__hip_atomic_load(&bar[1], __ATOMIC_ACQUIRE, __HIP_MEMORY_SCOPE_AGENT) == gen0)
                __builtin_amdgcn_s_sleep(4);
        }
    }
    __syncthreads();
}

// ---------------- phase 2: persistent recurrent scan ----------------
// 16 WGs x 8 waves = 128 waves; wave = (hcol 0..63, mHalf 0..1).
// h (64x1024 bf16) is staged into padded LDS per WG per step; A fragments come
// from ds_load_b128. Each wave owns the i/f/g/o tiles of 2 M-tiles, so each
// W_hh fragment feeds 2 WMMA and the cell math fuses in-register.
__global__ __launch_bounds__(256) void lstm_scan_kernel(const unsigned short* __restrict__ whh,
                                                        const float* __restrict__ xg,
                                                        float* __restrict__ y,
                                                        unsigned short* __restrict__ hbf, // 2 x [64*1024]
                                                        float* __restrict__ c,            // [64*1024]
                                                        unsigned* __restrict__ bar) {
    extern __shared__ unsigned short hlds[];                 // 64 x LDS_PITCH shorts
    const int lane  = threadIdx.x & 31;
    const int wv    = blockIdx.x * 8 + (threadIdx.x >> 5);   // 0..127
    const int hcol  = wv >> 1;                               // 0..63
    const int mHalf = wv & 1;                                // 0..1 (rows mHalf*32..+31)
    const int colLane = lane & 15;
    const int kh      = lane >> 4;
    const int n       = hcol * 16 + colLane;
    const int row0    = mHalf * 32;

    // cooperative zero-init of h (both buffers) and c
    const int tid = blockIdx.x * blockDim.x + threadIdx.x;
    for (int i = tid; i < NB * NHID; i += NWG2 * 256) {
        hbf[i] = 0;
        hbf[NB * NHID + i] = 0;
        c[i] = 0.0f;
    }
    grid_barrier(bar, NWG2);

    for (int t = 0; t < T_SEQ; ++t) {
        const unsigned short* hcur = hbf + (size_t)(t & 1) * (NB * NHID);
        unsigned short* hnext      = hbf + (size_t)((t + 1) & 1) * (NB * NHID);

        // stage h into LDS (8192 x 16B chunks, padded pitch)
        for (int i = threadIdx.x; i < (NB * NHID) / 8; i += 256) {
            const int row = i >> 7;          // 128 chunks per 1024-short row
            const int cir = i & 127;
            *reinterpret_cast<v8bf*>(hlds + row * LDS_PITCH + cir * 8) =
                *reinterpret_cast<const v8bf*>(hcur + row * 1024 + cir * 8);
        }
        __syncthreads();

        v8f acc[2][4];
#pragma unroll
        for (int s = 0; s < 2; ++s)
#pragma unroll
            for (int g = 0; g < 4; ++g) acc[s][g] = zero8();

        for (int kk = 0; kk < NHID; kk += 32) {
            v16bf a0 = load_frag_A_lds(hlds, row0,      kk, lane);
            v16bf a1 = load_frag_A_lds(hlds, row0 + 16, kk, lane);
#pragma unroll
            for (int g = 0; g < 4; ++g) {
                v16bf b = load_frag_B(whh, NHID, g * NHID + hcol * 16, kk, lane);
                acc[0][g] = __builtin_amdgcn_wmma_f32_16x16x32_bf16(false, a0, false, b, (short)0, acc[0][g], false, false);
                acc[1][g] = __builtin_amdgcn_wmma_f32_16x16x32_bf16(false, a1, false, b, (short)0, acc[1][g], false, false);
            }
        }

        const float* xgt = xg + (size_t)t * NB * G4;
#pragma unroll
        for (int s = 0; s < 2; ++s) {
#pragma unroll
            for (int r = 0; r < 8; ++r) {
                const int m = row0 + s * 16 + r + kh * 8;    // batch row 0..63
                const float* xr = xgt + (size_t)m * G4;
                const float gi = acc[s][0][r] + xr[0 * NHID + n];
                const float gf = acc[s][1][r] + xr[1 * NHID + n];
                const float gg = acc[s][2][r] + xr[2 * NHID + n];
                const float go = acc[s][3][r] + xr[3 * NHID + n];
                const float cn = sigmoid_f(gf) * c[m * NHID + n] + sigmoid_f(gi) * tanh_f(gg);
                c[m * NHID + n] = cn;
                const float hn = sigmoid_f(go) * tanh_f(cn);
                y[((size_t)t * NB + m) * NHID + n] = hn;
                hnext[m * NHID + n] = f2bf(hn);
            }
        }
        // barrier: h writes visible to all WGs; its __syncthreads also
        // protects the LDS image before next step's re-fill
        grid_barrier(bar, NWG2);
    }
}

// ---------------- launch ----------------
extern "C" void kernel_launch(void* const* d_in, const int* in_sizes, int n_in,
                              void* d_out, int out_size, void* d_ws, size_t ws_size,
                              hipStream_t stream) {
    const float* x    = (const float*)d_in[0];   // [512,64,1024]
    const float* W_ih = (const float*)d_in[1];   // [4096,1024]
    const float* W_hh = (const float*)d_in[2];   // [4096,1024]
    const float* b_ih = (const float*)d_in[3];   // [4096]
    const float* b_hh = (const float*)d_in[4];   // [4096]
    float* y = (float*)d_out;                    // [512,64,1024]

    // workspace carve-up (all 256B aligned)
    char* ws = (char*)d_ws;
    const size_t XBF   = (size_t)T_SEQ * NB * NIN * 2;   //  64 MB
    const size_t WBF   = (size_t)G4 * NIN * 2;           //   8 MB each
    const size_t XG    = (size_t)T_SEQ * NB * G4 * 4;    // 512 MB
    const size_t HBF   = (size_t)2 * NB * NHID * 2;      // 256 KB (ping-pong)
    const size_t CBUF  = (size_t)NB * NHID * 4;          // 256 KB
    unsigned short* xbf   = (unsigned short*)(ws);
    unsigned short* wihbf = (unsigned short*)(ws + XBF);
    unsigned short* whhbf = (unsigned short*)(ws + XBF + WBF);
    float*          xg    = (float*)(ws + XBF + 2 * WBF);
    unsigned short* hbf   = (unsigned short*)(ws + XBF + 2 * WBF + XG);
    float*          cbuf  = (float*)(ws + XBF + 2 * WBF + XG + HBF);
    unsigned*       bar   = (unsigned*)(ws + XBF + 2 * WBF + XG + HBF + CBUF);

    // barrier words must start at 0 every launch (ws is poisoned once)
    hipMemsetAsync(bar, 0, 256, stream);

    // f32 -> bf16 conversions
    {
        int n8 = (T_SEQ * NB * NIN) / 8;
        cvt_f32_bf16<<<(n8 + 255) / 256, 256, 0, stream>>>(x, xbf, n8);
        n8 = (G4 * NIN) / 8;
        cvt_f32_bf16<<<(n8 + 255) / 256, 256, 0, stream>>>(W_ih, wihbf, n8);
        cvt_f32_bf16<<<(n8 + 255) / 256, 256, 0, stream>>>(W_hh, whhbf, n8);
    }

    // phase 1: big GEMM (1024 M-blocks of 32 rows * 64 N-groups / 8 waves per block)
    xgemm_kernel<<<8192, 256, 0, stream>>>(xbf, wihbf, b_ih, b_hh, xg);

    // phase 2: persistent recurrent scan with LDS-staged h
    const size_t ldsBytes = (size_t)NB * LDS_PITCH * sizeof(unsigned short); // 132096 B
    lstm_scan_kernel<<<NWG2, 256, ldsBytes, stream>>>(whhbf, xg, y, hbf, cbuf, bar);
}
